// CorticalBlock_59923383714135
// MI455X (gfx1250) — matrix operand order repaired and checked by reading, not compiled, not run on hardware
//
#include <hip/hip_runtime.h>
#include <hip/hip_bf16.h>
#include <math.h>

typedef __bf16 bf16;
typedef __attribute__((ext_vector_type(16))) __bf16 v16bf;
typedef __attribute__((ext_vector_type(8)))  __bf16 v8bf;
typedef __attribute__((ext_vector_type(8)))  float  v8f;

#define WMMA_BF16(A_, B_, C_) \
  __builtin_amdgcn_wmma_f32_16x16x32_bf16(false, (A_), false, (B_), (short)0, (C_), false, false)

// ---------- fp32 <-> bf16 helpers (round-to-nearest-even) ----------
__device__ __forceinline__ bf16 f2bf(float f) {
  unsigned u = __builtin_bit_cast(unsigned, f);
  unsigned r = u + 0x7FFFu + ((u >> 16) & 1u);
  unsigned short h = (unsigned short)(r >> 16);
  return __builtin_bit_cast(bf16, h);
}
__device__ __forceinline__ float bf2f(bf16 b) {
  unsigned short h = __builtin_bit_cast(unsigned short, b);
  unsigned u = ((unsigned)h) << 16;
  return __builtin_bit_cast(float, u);
}

// ---------- WMMA fragment loaders (ISA 7.12.2 layouts) ----------
// A fragment: 16x32 bf16. lane holds row (lane&15); K chunks [klo..klo+7] and
// [klo+16..klo+23], klo = (lane<16 ? 0 : 8). Two 16B loads per lane.
__device__ __forceinline__ v16bf load_fragA(const bf16* __restrict__ base, int ld, int lane) {
  const bf16* p = base + (size_t)(lane & 15) * ld + ((lane & 16) ? 8 : 0);
  v8bf lo = *(const v8bf*)(p);
  v8bf hi = *(const v8bf*)(p + 16);
  return __builtin_shufflevector(lo, hi, 0,1,2,3,4,5,6,7,8,9,10,11,12,13,14,15);
}
// B fragment: 32x16 bf16 loaded from B^T stored row-major (N x K). lane holds
// column N=(lane&15); K chunks [klo..klo+7] and [klo+16..klo+23], klo = (lane<16 ? 0 : 8).
__device__ __forceinline__ v16bf load_fragB(const bf16* __restrict__ base, int ld, int lane) {
  const bf16* p = base + (size_t)(lane & 15) * ld + ((lane & 16) ? 8 : 0);
  v8bf lo = *(const v8bf*)(p);
  v8bf hi = *(const v8bf*)(p + 16);
  return __builtin_shufflevector(lo, hi, 0,1,2,3,4,5,6,7,8,9,10,11,12,13,14,15);
}

// ---------- generic WMMA GEMM, 2-stage software pipeline, fused epilogues ----------
// C[M,N] (+epilogue) = A[M,K](bf16, lda) @ Wt[N,K]^T (bf16, ldw)
// EPI: 0=f32 store, 1=bf16 store, 2=thalamic gate, 3=1+tanh, 4=mul-by-aux bf16, 5=sigmoid f32
template <int EPI>
__global__ __launch_bounds__(256) void gemm_wmma(
    const bf16* __restrict__ A, int lda,
    const bf16* __restrict__ Wt, int ldw,
    void* __restrict__ Cout,
    int M, int N, int K, float scale,
    const float* __restrict__ e0, const float* __restrict__ e1) {
  __shared__ __align__(16) bf16 As[2][128][40];
  __shared__ __align__(16) bf16 Bs[2][64][40];
  const int tid = threadIdx.x, wave = tid >> 5, lane = tid & 31;
  const int m0 = blockIdx.y * 128, n0 = blockIdx.x * 64;
  const int wm = wave & 3, wn = wave >> 2;  // 4 waves along M x 2 along N
  const int row0 = tid >> 2, ch = tid & 3;  // staging assignment
  v8f acc[2][2] = {};

  // prologue: stage K-tile 0
  v8bf a_c0 = *(const v8bf*)(A + (size_t)(m0 + row0) * lda + ch * 8);
  v8bf a_c1 = *(const v8bf*)(A + (size_t)(m0 + row0 + 64) * lda + ch * 8);
  v8bf b_c  = *(const v8bf*)(Wt + (size_t)(n0 + row0) * ldw + ch * 8);
  *(v8bf*)&As[0][row0][ch * 8] = a_c0;
  *(v8bf*)&As[0][row0 + 64][ch * 8] = a_c1;
  *(v8bf*)&Bs[0][row0][ch * 8] = b_c;
  __syncthreads();

  int cur = 0;
  for (int kb = 0; kb < K; kb += 32) {
    const bool more = (kb + 32) < K;
    if (more) {  // issue next tile's global loads; latency hidden behind WMMAs
      a_c0 = *(const v8bf*)(A + (size_t)(m0 + row0) * lda + kb + 32 + ch * 8);
      a_c1 = *(const v8bf*)(A + (size_t)(m0 + row0 + 64) * lda + kb + 32 + ch * 8);
      b_c  = *(const v8bf*)(Wt + (size_t)(n0 + row0) * ldw + kb + 32 + ch * 8);
    }
    v16bf a0 = load_fragA(&As[cur][wm * 32][0], 40, lane);
    v16bf a1 = load_fragA(&As[cur][wm * 32 + 16][0], 40, lane);
    v16bf b0 = load_fragB(&Bs[cur][wn * 32][0], 40, lane);
    v16bf b1 = load_fragB(&Bs[cur][wn * 32 + 16][0], 40, lane);
    acc[0][0] = WMMA_BF16(a0, b0, acc[0][0]);
    acc[0][1] = WMMA_BF16(a0, b1, acc[0][1]);
    acc[1][0] = WMMA_BF16(a1, b0, acc[1][0]);
    acc[1][1] = WMMA_BF16(a1, b1, acc[1][1]);
    if (more) {
      *(v8bf*)&As[cur ^ 1][row0][ch * 8] = a_c0;
      *(v8bf*)&As[cur ^ 1][row0 + 64][ch * 8] = a_c1;
      *(v8bf*)&Bs[cur ^ 1][row0][ch * 8] = b_c;
    }
    __syncthreads();
    cur ^= 1;
  }

#pragma unroll
  for (int i = 0; i < 2; ++i)
#pragma unroll
    for (int j = 0; j < 2; ++j)
#pragma unroll
      for (int r = 0; r < 8; ++r) {
        int m = m0 + wm * 32 + i * 16 + r + ((lane & 16) ? 8 : 0);
        int n = n0 + wn * 32 + j * 16 + (lane & 15);
        float v = acc[i][j][r] * scale;
        size_t idx = (size_t)m * N + n;
        if constexpr (EPI == 0) {
          ((float*)Cout)[idx] = v;
        } else if constexpr (EPI == 1) {
          ((bf16*)Cout)[idx] = f2bf(v);
        } else if constexpr (EPI == 2) {  // thalamic gate: x * (ACH*sig(v+bg)+1-ACH)
          float g = 1.f / (1.f + __expf(-(v + e0[n])));
          ((float*)Cout)[idx] = e1[idx] * (0.5f * g + 0.5f);
        } else if constexpr (EPI == 3) {  // apical factor
          ((float*)Cout)[idx] = 1.f + tanhf(v);
        } else if constexpr (EPI == 4) {  // attn_out@Wo * factor
          ((bf16*)Cout)[idx] = f2bf(v * e0[idx]);
        } else if constexpr (EPI == 5) {  // receptance
          ((float*)Cout)[idx] = 1.f / (1.f + __expf(-v));
        }
      }
}

// ---------- fp32 weight -> bf16 transposed (Wt[n*ldo + koff + k] = W[k][n]) ----------
__global__ __launch_bounds__(256) void wtrans(const float* __restrict__ W,
                                              bf16* __restrict__ Wt,
                                              int K, int N, int ldo, int koff) {
  __shared__ float tile[32][33];
  int kb = blockIdx.x * 32, nb = blockIdx.y * 32;
  int tx = threadIdx.x, ty = threadIdx.y;
  for (int i = ty; i < 32; i += 8) tile[i][tx] = W[(size_t)(kb + i) * N + nb + tx];
  __syncthreads();
  for (int i = ty; i < 32; i += 8)
    Wt[(size_t)(nb + i) * ldo + koff + kb + tx] = f2bf(tile[tx][i]);
}

// ---------- x / context -> concatenated bf16 activations (M x 2D) ----------
__global__ __launch_bounds__(256) void xcconv(const float* __restrict__ x,
                                              const float* __restrict__ ctx,
                                              bf16* __restrict__ xc, int M, int Dd) {
  int i = blockIdx.x * 256 + threadIdx.x;
  if (i < M * Dd) {
    int m = i / Dd, c = i % Dd;
    xc[(size_t)m * 2 * Dd + c] = f2bf(x[i]);
    xc[(size_t)m * 2 * Dd + Dd + c] = f2bf(ctx[i]);
  }
}

// ---------- per-batch bf16 transpose: vT[b][d][s] = v[b][s][d] ----------
__global__ __launch_bounds__(256) void vtrans(const bf16* __restrict__ v,
                                              bf16* __restrict__ vT, int T, int D) {
  __shared__ bf16 tile[32][33];
  int b = blockIdx.z, sb = blockIdx.x * 32, db = blockIdx.y * 32;
  int tx = threadIdx.x, ty = threadIdx.y;
  const bf16* vb = v + (size_t)b * T * D;
  bf16* ob = vT + (size_t)b * D * T;
  for (int i = ty; i < 32; i += 8) tile[i][tx] = vb[(size_t)(sb + i) * D + db + tx];
  __syncthreads();
  for (int i = ty; i < 32; i += 8) ob[(size_t)(db + i) * T + sb + tx] = tile[tx][i];
}

// ---------- LIF scans (sequential over T; prefetch ahead on the stride-D stream) ----------
__global__ __launch_bounds__(256) void lif1(const float* __restrict__ g,
                                            bf16* __restrict__ s, int T, int D) {
  int ch = blockIdx.x * 256 + threadIdx.x;
  int b = ch / D, d = ch % D;
  const float* gp = g + (size_t)b * T * D + d;
  bf16* sp = s + (size_t)b * T * D + d;
  float u = 0.f;
  for (int t = 0; t < T; ++t) {
    if (t + 8 < T) __builtin_prefetch(gp + (size_t)(t + 8) * D, 0, 3);
    u = 0.9f * u + gp[(size_t)t * D];
    float spk = (u >= 1.0f) ? 1.f : 0.f;
    sp[(size_t)t * D] = f2bf(spk);
    u -= spk;  // soft reset (VTH = 1)
  }
}
__global__ __launch_bounds__(256) void lif2(const float* __restrict__ rec,
                                            const bf16* __restrict__ spk,
                                            float* __restrict__ y, int T, int D) {
  int ch = blockIdx.x * 256 + threadIdx.x;
  int b = ch / D, d = ch % D;
  const float* rp = rec + (size_t)b * T * D + d;
  const bf16* sp = spk + (size_t)b * T * D + d;
  float* yp = y + (size_t)b * T * D + d;
  float u = 0.f;
  for (int t = 0; t < T; ++t) {
    if (t + 8 < T) __builtin_prefetch(rp + (size_t)(t + 8) * D, 0, 3);
    u = 0.9f * u + rp[(size_t)t * D];
    float s = (u >= 1.0f) ? 1.f : 0.f;
    yp[(size_t)t * D] = s + bf2f(sp[(size_t)t * D]);
    u -= s;
  }
}

// ---------- numerically-stable RWKV WKV scan; writes rw = r * wkv (bf16) ----------
__global__ __launch_bounds__(256) void wkvk(const float* __restrict__ rk,
                                            const float* __restrict__ rv,
                                            const float* __restrict__ rr,
                                            const float* __restrict__ w_decay,
                                            const float* __restrict__ u_first,
                                            bf16* __restrict__ rw, int T, int D) {
  int ch = blockIdx.x * 256 + threadIdx.x;
  int b = ch / D, d = ch % D;
  float w = __expf(w_decay[d]);
  float uf = u_first[d];
  float aa = 0.f, bb = 0.f, pp = -1e30f;
  size_t base = (size_t)b * T * D + d;
  for (int t = 0; t < T; ++t) {
    size_t idx = base + (size_t)t * D;
    if (t + 8 < T) {
      __builtin_prefetch(rk + idx + 8 * (size_t)D, 0, 3);
      __builtin_prefetch(rv + idx + 8 * (size_t)D, 0, 3);
      __builtin_prefetch(rr + idx + 8 * (size_t)D, 0, 3);
    }
    float kt = rk[idx], vt = rv[idx];
    float ww = uf + kt;
    float p = fmaxf(pp, ww);
    float e1 = __expf(pp - p), e2 = __expf(ww - p);
    float wkv = (e1 * aa + e2 * vt) / (e1 * bb + e2);
    rw[idx] = f2bf(rr[idx] * wkv);
    float ww2 = pp - w;
    float p2 = fmaxf(ww2, kt);
    e1 = __expf(ww2 - p2);
    e2 = __expf(kt - p2);
    aa = e1 * aa + e2 * vt;
    bb = e1 * bb + e2;
    pp = p2;
  }
}

// ---------- flash-style spike attention (dh=128), q pre-scaled by 1/sqrt(dh) ----------
__global__ __launch_bounds__(256) void attn_kernel(const bf16* __restrict__ q,
                                                   const bf16* __restrict__ k,
                                                   const bf16* __restrict__ vT,
                                                   bf16* __restrict__ out, int T, int D) {
  const int dh = 128;
  int b = blockIdx.z, h = blockIdx.y;
  int wave = threadIdx.x >> 5, lane = threadIdx.x & 31;
  int q0 = blockIdx.x * 128 + wave * 16;
  const bf16* qbase = q + (size_t)(b * T + q0) * D + h * dh;
  const bf16* kbase = k + (size_t)b * T * D + h * dh;
  const bf16* vtb = vT + ((size_t)b * D + (size_t)h * dh) * T;
  __shared__ __align__(16) bf16 psc[8][16][40];  // per-wave P scratch (16x32)

  v16bf qf[4];
#pragma unroll
  for (int c = 0; c < 4; ++c) qf[c] = load_fragA(qbase + c * 32, D, lane);

  v8f o[8] = {};
  float mrun[8], lsum[8];
#pragma unroll
  for (int r = 0; r < 8; ++r) { mrun[r] = -1e30f; lsum[r] = 0.f; }

  for (int s0 = 0; s0 < T; s0 += 32) {
    // hoist ALL fragment loads for this key block: one clause, one wait,
    // then back-to-back WMMAs (instead of load->wait->wmma chains)
    v16bf kf[8], vf[8];
#pragma unroll
    for (int c = 0; c < 4; ++c) {
      kf[c]     = load_fragB(kbase + (size_t)s0 * D + c * 32, D, lane);
      kf[c + 4] = load_fragB(kbase + (size_t)(s0 + 16) * D + c * 32, D, lane);
    }
#pragma unroll
    for (int n0 = 0; n0 < 8; ++n0)
      vf[n0] = load_fragB(vtb + (size_t)(n0 * 16) * T + s0, T, lane);

    v8f S0 = {}, S1 = {};
#pragma unroll
    for (int c = 0; c < 4; ++c) {
      S0 = WMMA_BF16(qf[c], kf[c], S0);
      S1 = WMMA_BF16(qf[c], kf[c + 4], S1);
    }
    // online softmax per query row (row spread over 16 lanes of one half-wave)
#pragma unroll
    for (int r = 0; r < 8; ++r) {
      float mx = fmaxf(S0[r], S1[r]);
      mx = fmaxf(mx, __shfl_xor(mx, 1, 32));
      mx = fmaxf(mx, __shfl_xor(mx, 2, 32));
      mx = fmaxf(mx, __shfl_xor(mx, 4, 32));
      mx = fmaxf(mx, __shfl_xor(mx, 8, 32));
      float nm = fmaxf(mrun[r], mx);
      float corr = __expf(mrun[r] - nm);
      mrun[r] = nm;
      float p0 = __expf(S0[r] - nm), p1 = __expf(S1[r] - nm);
      float ps = p0 + p1;
      ps += __shfl_xor(ps, 1, 32);
      ps += __shfl_xor(ps, 2, 32);
      ps += __shfl_xor(ps, 4, 32);
      ps += __shfl_xor(ps, 8, 32);
      lsum[r] = lsum[r] * corr + ps;
#pragma unroll
      for (int n0 = 0; n0 < 8; ++n0) o[n0][r] *= corr;
      int mrow = r + ((lane & 16) ? 8 : 0);
      psc[wave][mrow][lane & 15] = f2bf(p0);
      psc[wave][mrow][(lane & 15) + 16] = f2bf(p1);
    }
    // O += P (16x32) @ V (32x16 per dh tile); V fragments already in registers
    v16bf pA = load_fragA(&psc[wave][0][0], 40, lane);
#pragma unroll
    for (int n0 = 0; n0 < 8; ++n0) o[n0] = WMMA_BF16(pA, vf[n0], o[n0]);
  }
  bf16* ob = out + (size_t)b * T * D + h * dh;
#pragma unroll
  for (int n0 = 0; n0 < 8; ++n0)
#pragma unroll
    for (int r = 0; r < 8; ++r) {
      int m = q0 + r + ((lane & 16) ? 8 : 0);
      int n = n0 * 16 + (lane & 15);
      ob[(size_t)m * D + n] = f2bf(o[n0][r] / lsum[r]);
    }
}

// ---------- LayerNorm over D (in place) ----------
__global__ __launch_bounds__(256) void ln_kernel(float* __restrict__ y,
                                                 const float* __restrict__ gamma,
                                                 const float* __restrict__ beta, int D) {
  int row = blockIdx.x;
  float* p = y + (size_t)row * D;
  float s = 0.f, s2 = 0.f;
  for (int i = threadIdx.x; i < D; i += 256) { float v = p[i]; s += v; s2 += v * v; }
  for (int m = 16; m >= 1; m >>= 1) { s += __shfl_xor(s, m, 32); s2 += __shfl_xor(s2, m, 32); }
  __shared__ float sh[16];
  int wave = threadIdx.x >> 5, lane = threadIdx.x & 31;
  if (lane == 0) { sh[wave] = s; sh[wave + 8] = s2; }
  __syncthreads();
  if (threadIdx.x == 0) {
    float a = 0.f, b2 = 0.f;
    for (int i = 0; i < 8; ++i) { a += sh[i]; b2 += sh[i + 8]; }
    sh[0] = a; sh[8] = b2;
  }
  __syncthreads();
  float mean = sh[0] / (float)D;
  float var = sh[8] / (float)D - mean * mean;
  float rstd = rsqrtf(var + 1e-5f);
  for (int i = threadIdx.x; i < D; i += 256)
    p[i] = (p[i] - mean) * rstd * gamma[i] + beta[i];
}

// =====================================================================
extern "C" void kernel_launch(void* const* d_in, const int* in_sizes, int n_in,
                              void* d_out, int out_size, void* d_ws, size_t ws_size,
                              hipStream_t stream) {
  (void)in_sizes; (void)n_in; (void)out_size; (void)ws_size;
  const int B = 4, T = 1024, D = 1024, M = B * T, K2 = 2 * D;

  const float* x   = (const float*)d_in[0];
  const float* ctx = (const float*)d_in[1];
  const float* Wg  = (const float*)d_in[2];
  const float* bg  = (const float*)d_in[3];
  const float* Wc  = (const float*)d_in[4];
  const float* Wq  = (const float*)d_in[5];
  const float* Wk  = (const float*)d_in[6];
  const float* Wv  = (const float*)d_in[7];
  const float* Wo  = (const float*)d_in[8];
  const float* Wm  = (const float*)d_in[9];
  const float* rWk = (const float*)d_in[10];
  const float* rWv = (const float*)d_in[11];
  const float* rWr = (const float*)d_in[12];
  const float* rWo = (const float*)d_in[13];
  const float* w_decay = (const float*)d_in[14];
  const float* u_first = (const float*)d_in[15];
  const float* gamma = (const float*)d_in[16];
  const float* beta  = (const float*)d_in[17];
  float* out = (float*)d_out;

  // workspace arena (~142 MB), lifetime-safe aliasing
  char* basep = (char*)d_ws;
  size_t off = 0;
  auto alloc = [&](size_t bytes) -> char* {
    char* p = basep + off;
    off = (off + bytes + 255) & ~(size_t)255;
    return p;
  };
  bf16* Wgc  = (bf16*)alloc((size_t)D * K2 * 2);  // [Wg;Wc]^T, N x 2K
  bf16* WtQ  = (bf16*)alloc((size_t)D * D * 2);
  bf16* WtK  = (bf16*)alloc((size_t)D * D * 2);
  bf16* WtV  = (bf16*)alloc((size_t)D * D * 2);
  bf16* WtO  = (bf16*)alloc((size_t)D * D * 2);
  bf16* WtM  = (bf16*)alloc((size_t)D * D * 2);
  bf16* WtRK = (bf16*)alloc((size_t)D * D * 2);
  bf16* WtRV = (bf16*)alloc((size_t)D * D * 2);
  bf16* WtRR = (bf16*)alloc((size_t)D * D * 2);
  bf16* WtRO = (bf16*)alloc((size_t)D * D * 2);
  bf16* xc    = (bf16*)alloc((size_t)M * K2 * 2);   // [x | ctx] bf16
  bf16* spk   = (bf16*)alloc((size_t)M * D * 2);    // spikes2 (live to end)
  float* gated = (float*)alloc((size_t)M * D * 4);  // -> later rk
  float* tanhF = (float*)alloc((size_t)M * D * 4);  // -> later rv
  float* rrb   = (float*)alloc((size_t)M * D * 4);  // receptance
  float* rec   = (float*)alloc((size_t)M * D * 4);  // recurrent
  bf16* qb   = (bf16*)alloc((size_t)M * D * 2);     // -> later rw
  bf16* kbuf = (bf16*)alloc((size_t)M * D * 2);     // -> later modulated
  bf16* vbuf = (bf16*)alloc((size_t)M * D * 2);     // -> later attnOut
  bf16* vTb  = (bf16*)alloc((size_t)M * D * 2);
  float* rkb = gated;   // gated dead after lif1
  float* rvb = tanhF;   // tanhF dead after modulate GEMM
  bf16* rwb = qb;       // q dead after attention
  bf16* modb = kbuf;    // k dead after attention
  bf16* attnO = vbuf;   // v dead after vtrans

  dim3 tb32(32, 8);
  dim3 gemmGrid(D / 64, M / 128);

  // weights -> bf16 transposed
  wtrans<<<dim3(32, 32), tb32, 0, stream>>>(Wg, Wgc, D, D, K2, 0);
  wtrans<<<dim3(32, 32), tb32, 0, stream>>>(Wc, Wgc, D, D, K2, D);
  wtrans<<<dim3(32, 32), tb32, 0, stream>>>(Wq, WtQ, D, D, D, 0);
  wtrans<<<dim3(32, 32), tb32, 0, stream>>>(Wk, WtK, D, D, D, 0);
  wtrans<<<dim3(32, 32), tb32, 0, stream>>>(Wv, WtV, D, D, D, 0);
  wtrans<<<dim3(32, 32), tb32, 0, stream>>>(Wo, WtO, D, D, D, 0);
  wtrans<<<dim3(32, 32), tb32, 0, stream>>>(Wm, WtM, D, D, D, 0);
  wtrans<<<dim3(32, 32), tb32, 0, stream>>>(rWk, WtRK, D, D, D, 0);
  wtrans<<<dim3(32, 32), tb32, 0, stream>>>(rWv, WtRV, D, D, D, 0);
  wtrans<<<dim3(32, 32), tb32, 0, stream>>>(rWr, WtRR, D, D, D, 0);
  wtrans<<<dim3(32, 32), tb32, 0, stream>>>(rWo, WtRO, D, D, D, 0);
  xcconv<<<(M * D) / 256, 256, 0, stream>>>(x, ctx, xc, M, D);

  // stage 1: gate GEMM (fused K=2048: x@Wg + ctx@Wc), epilogue -> gated fp32
  gemm_wmma<2><<<gemmGrid, 256, 0, stream>>>(xc, K2, Wgc, K2, gated, M, D, K2, 1.f, bg, x);
  // stage 2: LIF scan
  lif1<<<16, 256, 0, stream>>>(gated, spk, T, D);
  // stage 3: q,k,v projections (q pre-scaled by 1/sqrt(dh))
  gemm_wmma<1><<<gemmGrid, 256, 0, stream>>>(spk, D, WtQ, D, qb, M, D, D,
                                             0.08838834764831845f, nullptr, nullptr);
  gemm_wmma<1><<<gemmGrid, 256, 0, stream>>>(spk, D, WtK, D, kbuf, M, D, D, 1.f, nullptr, nullptr);
  gemm_wmma<1><<<gemmGrid, 256, 0, stream>>>(spk, D, WtV, D, vbuf, M, D, D, 1.f, nullptr, nullptr);
  vtrans<<<dim3(32, 32, B), tb32, 0, stream>>>(vbuf, vTb, T, D);
  attn_kernel<<<dim3(T / 128, 8, B), 256, 0, stream>>>(qb, kbuf, vTb, attnO, T, D);
  // stage 4: apical modulation factor and modulated = (attn@Wo) * (1+tanh(ctx@Wm))
  gemm_wmma<3><<<gemmGrid, 256, 0, stream>>>(xc + D, K2, WtM, D, tanhF, M, D, D, 1.f,
                                             nullptr, nullptr);
  gemm_wmma<4><<<gemmGrid, 256, 0, stream>>>(attnO, D, WtO, D, modb, M, D, D, 1.f,
                                             tanhF, nullptr);
  // stage 5: RWKV projections + scan + output projection
  gemm_wmma<0><<<gemmGrid, 256, 0, stream>>>(modb, D, WtRK, D, rkb, M, D, D, 1.f, nullptr, nullptr);
  gemm_wmma<0><<<gemmGrid, 256, 0, stream>>>(modb, D, WtRV, D, rvb, M, D, D, 1.f, nullptr, nullptr);
  gemm_wmma<5><<<gemmGrid, 256, 0, stream>>>(modb, D, WtRR, D, rrb, M, D, D, 1.f, nullptr, nullptr);
  wkvk<<<16, 256, 0, stream>>>(rkb, rvb, rrb, w_decay, u_first, rwb, T, D);
  gemm_wmma<0><<<gemmGrid, 256, 0, stream>>>(rwb, D, WtRO, D, rec, M, D, D, 1.f, nullptr, nullptr);
  // stage 6: output LIF + residual, then LayerNorm in place on d_out
  lif2<<<16, 256, 0, stream>>>(rec, spk, out, T, D);
  ln_kernel<<<M, 256, 0, stream>>>(out, gamma, beta, D);
}
